// GATLayer_11330123727478
// MI455X (gfx1250) — matrix operand order repaired and checked by reading, not compiled, not run on hardware
//
#include <hip/hip_runtime.h>
#include <hip/hip_bf16.h>
#include <math.h>

// ---------------------------------------------------------------------------
// GAT layer, fused flash-attention style, CDNA5 (gfx1250, wave32, WMMA bf16,
// async global->LDS staging). N=4096, F_in=512, H=8, F_out=64.
//
// Pipeline (all scratch in d_ws, no atomics -> deterministic):
//   1. pack_h/pack_w : f32 -> bf16 WMMA A/B register layouts
//   2. hw_gemm       : hw[h] = h @ W[h]          (v_wmma_f32_16x16x32_bf16)
//   3. pack_hw, f1f2
//   4. attn          : 8-wave blocks; per 32-key chunk the packed hw B tile
//                      (4 KB) is staged in LDS with GLOBAL_LOAD_ASYNC_TO_LDS
//                      (double buffered, overlapped with softmax VALU work),
//                      online-softmax + WMMA accumulate, elu epilogue
//   5. head_mean
// ---------------------------------------------------------------------------

typedef __attribute__((ext_vector_type(16))) __bf16        v16bf;
typedef __attribute__((ext_vector_type(8)))  float         v8f;
typedef __attribute__((ext_vector_type(8)))  unsigned int  v8u;

#define NN      4096
#define FIN     512
#define HH      8
#define OO      64
#define GAT_ALPHA  0.2f
#define GAT_MASK   -9.0e15f

// ---- workspace layout (bytes) ---------------------------------------------
static const size_t OFF_HW    = 0;                                                  // 8 MB f32 hw
static const size_t OFF_APACK = OFF_HW    + (size_t)HH * NN * OO * 4;               // 4 MB
static const size_t OFF_WPACK = OFF_APACK + (size_t)(NN / 16) * (FIN / 32) * 32 * 8 * 4;  // 0.5 MB
static const size_t OFF_HWPK  = OFF_WPACK + (size_t)HH * (FIN / 32) * 4 * 32 * 8 * 4;     // 4 MB
static const size_t OFF_F1    = OFF_HWPK  + (size_t)HH * (NN / 32) * 4 * 32 * 8 * 4;
static const size_t OFF_F2    = OFF_F1    + (size_t)HH * NN * 4;
static const size_t OFF_HP    = OFF_F2    + (size_t)HH * NN * 4;                    // 8 MB

// round-to-nearest-even f32 -> bf16, packed pair (lo -> [15:0], hi -> [31:16])
__device__ __forceinline__ unsigned int pk_bf16(float lo, float hi) {
  unsigned int ul = __float_as_uint(lo);
  unsigned int uh = __float_as_uint(hi);
  ul = (ul + 0x7FFFu + ((ul >> 16) & 1u)) >> 16;
  uh = (uh + 0x7FFFu + ((uh >> 16) & 1u)) >> 16;
  return ul | (uh << 16);
}

// A-matrix (16x32, 16-bit) K offset for VGPR v, lane-half h (ISA 7.12.2)
__device__ __forceinline__ int kofsA(int v, int half) {
  return ((v < 4) ? (2 * v) : (16 + 2 * (v - 4))) + 8 * half;
}
// B-matrix (32x16, 16-bit): same K order as A (lanes 0-15 K=0..7,16..23; lanes 16-31 K=8..15,24..31)
__device__ __forceinline__ int kofsB(int v, int half) {
  return ((v < 4) ? (2 * v) : (16 + 2 * (v - 4))) + 8 * half;
}

// ---- 1a. pack h into A-matrix bf16 layout ---------------------------------
__global__ void __launch_bounds__(32)
gat_pack_h(const float* __restrict__ h, unsigned int* __restrict__ apack) {
  const int ks    = blockIdx.x;        // FIN/32 = 16
  const int ntile = blockIdx.y;        // N/16   = 256
  const int lane  = threadIdx.x;
  const int half  = lane >> 4;
  const int row   = ntile * 16 + (lane & 15);
  const float* src = h + (size_t)row * FIN + ks * 32;
  unsigned int* dst = apack + (((size_t)ntile * (FIN / 32) + ks) * 32 + lane) * 8;
#pragma unroll
  for (int v = 0; v < 8; ++v) {
    float2 t = *(const float2*)(src + kofsA(v, half));
    dst[v] = pk_bf16(t.x, t.y);
  }
}

// ---- 1b. pack W into B-matrix bf16 layout ---------------------------------
__global__ void __launch_bounds__(32)
gat_pack_w(const float* __restrict__ W, unsigned int* __restrict__ wpack) {
  const int ksc  = blockIdx.x;         // (FIN/32)*4 = 64
  const int head = blockIdx.y;         // 8
  const int ks = ksc >> 2, c = ksc & 3;
  const int lane = threadIdx.x;
  const int half = lane >> 4;
  const int col  = 16 * c + (lane & 15);
  unsigned int* dst = wpack + ((((size_t)head * (FIN / 32) + ks) * 4 + c) * 32 + lane) * 8;
#pragma unroll
  for (int v = 0; v < 8; ++v) {
    int k = ks * 32 + kofsB(v, half);
    float a = W[((size_t)head * FIN + k) * OO + col];
    float b = W[((size_t)head * FIN + k + 1) * OO + col];
    dst[v] = pk_bf16(a, b);
  }
}

// ---- 2. hw[h] = h @ W[h]  (16x64 tile per wave) ---------------------------
__global__ void __launch_bounds__(32)
gat_hw_gemm(const unsigned int* __restrict__ apack,
            const unsigned int* __restrict__ wpack,
            float* __restrict__ hw) {
  const int ntile = blockIdx.x;        // 256
  const int head  = blockIdx.y;        // 8
  const int lane  = threadIdx.x;
  v8f acc[4] = {};
  for (int ks = 0; ks < FIN / 32; ++ks) {
    // batch all loads first so they can overlap, then run the 4 WMMAs
    v8u araw = *(const v8u*)(apack + (((size_t)ntile * (FIN / 32) + ks) * 32 + lane) * 8);
    v8u braw[4];
#pragma unroll
    for (int c = 0; c < 4; ++c)
      braw[c] = *(const v8u*)(wpack + ((((size_t)head * (FIN / 32) + ks) * 4 + c) * 32 + lane) * 8);
    v16bf a = __builtin_bit_cast(v16bf, araw);
#pragma unroll
    for (int c = 0; c < 4; ++c)
      acc[c] = __builtin_amdgcn_wmma_f32_16x16x32_bf16(
          false, a, false, __builtin_bit_cast(v16bf, braw[c]), (short)0, acc[c], false, false);
  }
  const int half = lane >> 4, coll = lane & 15;
#pragma unroll
  for (int c = 0; c < 4; ++c)
#pragma unroll
    for (int v = 0; v < 8; ++v) {
      int row = ntile * 16 + v + 8 * half;
      hw[((size_t)head * NN + row) * OO + coll + 16 * c] = acc[c][v];
    }
}

// ---- 3a. pack hw into B-matrix bf16 layout --------------------------------
__global__ void __launch_bounds__(32)
gat_pack_hw(const float* __restrict__ hw, unsigned int* __restrict__ hwpack) {
  const int msc  = blockIdx.x;         // (N/32)*4 = 512
  const int head = blockIdx.y;
  const int ms = msc >> 2, c = msc & 3;
  const int lane = threadIdx.x;
  const int half = lane >> 4;
  const int col  = 16 * c + (lane & 15);
  unsigned int* dst = hwpack + ((((size_t)head * (NN / 32) + ms) * 4 + c) * 32 + lane) * 8;
#pragma unroll
  for (int v = 0; v < 8; ++v) {
    int m = ms * 32 + kofsB(v, half);
    float a = hw[((size_t)head * NN + m) * OO + col];
    float b = hw[((size_t)head * NN + m + 1) * OO + col];
    dst[v] = pk_bf16(a, b);
  }
}

// ---- 3b. f1/f2 projections ------------------------------------------------
__global__ void __launch_bounds__(256)
gat_f1f2(const float* __restrict__ hw, const float* __restrict__ a1,
         const float* __restrict__ a2, float* __restrict__ f1,
         float* __restrict__ f2) {
  const int t = blockIdx.x * blockDim.x + threadIdx.x;  // H*N
  const int head = t >> 12;
  const float* r  = hw + (size_t)t * OO;
  const float* p1 = a1 + head * OO;
  const float* p2 = a2 + head * OO;
  float s1 = 0.f, s2 = 0.f;
#pragma unroll 8
  for (int o = 0; o < OO; ++o) {
    float x = r[o];
    s1 += x * p1[o];
    s2 += x * p2[o];
  }
  f1[t] = s1;
  f2[t] = s2;
}

// issue one 16-byte async global->LDS copy (ASYNCcnt tracked)
__device__ __forceinline__ void async_copy16(unsigned int lds_addr,
                                             const unsigned int* gsrc) {
  asm volatile("global_load_async_to_lds_b128 %0, %1, off"
               :: "v"(lds_addr), "v"(gsrc) : "memory");
}
__device__ __forceinline__ void wait_async0() {
  asm volatile("s_wait_asynccnt 0x0" ::: "memory");
}

// ---- 4. fused masked-softmax attention + elu ------------------------------
// 8 waves / block. Each wave owns one 16-row query tile; the packed hw B
// chunk (4 KB / 32 keys) is shared by all waves and double-buffered in LDS
// via async copies that overlap the softmax VALU work of the current chunk.
__global__ void __launch_bounds__(256)
gat_attn(const float* __restrict__ f1, const float* __restrict__ f2,
         const int* __restrict__ adj, const unsigned int* __restrict__ hwpack,
         float* __restrict__ hprime) {
  __shared__ unsigned int ldsB[2][1024];   // [buf][c*32*8 + lane*8 + v]

  const int tid   = threadIdx.x;
  const int wave  = tid >> 5;
  const int lane  = tid & 31;
  const int head  = blockIdx.y;
  const int ntile = blockIdx.x * 8 + wave;   // grid.x = N/128 = 32
  const int half  = lane >> 4;
  const int lrow  = lane & 15;
  const int row   = ntile * 16 + lrow;

  const float myf1 = f1[head * NN + row];
  const float* f2h = f2 + head * NN;
  const int* adjrow = adj + (size_t)row * NN;
  const unsigned int* hwp = hwpack + (size_t)head * (NN / 32) * 1024;

  // prologue: stage chunk 0 into buffer 0 (one b128 per thread = 4 KB/block)
  async_copy16((unsigned int)(uintptr_t)&ldsB[0][tid * 4], hwp + tid * 4);
  wait_async0();
  __syncthreads();

  v8f acc[4] = {};
  float rm = -INFINITY, S = 0.f;

  for (int ms = 0; ms < NN / 32; ++ms) {
    const int cur = ms & 1;
    // kick off async staging of the next chunk into the other buffer;
    // it overlaps all the VALU + WMMA work below
    if (ms + 1 < NN / 32)
      async_copy16((unsigned int)(uintptr_t)&ldsB[cur ^ 1][tid * 4],
                   hwp + (size_t)(ms + 1) * 1024 + tid * 4);

    const int mbase = ms * 32;
    __builtin_prefetch(adjrow + mbase + 32, 0, 1);

    // build 16 e-values of this row in A-matrix K order
    float p[16];
    float lm = -INFINITY;
#pragma unroll
    for (int v = 0; v < 8; ++v) {
      int m = mbase + kofsA(v, half);
      float2 ff = *(const float2*)(f2h + m);
      int2   ad = *(const int2*)(adjrow + m);
      float e0 = myf1 + ff.x; e0 = (e0 > 0.f) ? e0 : GAT_ALPHA * e0;
      float e1 = myf1 + ff.y; e1 = (e1 > 0.f) ? e1 : GAT_ALPHA * e1;
      if (ad.x <= 0) e0 = GAT_MASK;
      if (ad.y <= 0) e1 = GAT_MASK;
      p[2 * v] = e0; p[2 * v + 1] = e1;
      lm = fmaxf(lm, fmaxf(e0, e1));
    }
    lm = fmaxf(lm, __shfl_xor(lm, 16, 32));     // row split across lanes L, L^16
    const float nm = fmaxf(rm, lm);
    const float sc = expf(rm - nm);             // 0 on first chunk (rm = -inf)
    float ls = 0.f;
#pragma unroll
    for (int i = 0; i < 16; ++i) { p[i] = expf(p[i] - nm); ls += p[i]; }
    ls += __shfl_xor(ls, 16, 32);
    S = S * sc + ls;
    rm = nm;

    // rescale accumulators: C row for VGPR v is (v + 8*half); scale lives in
    // lane (v + 8*half)
#pragma unroll
    for (int v = 0; v < 8; ++v) {
      float scv = __shfl(sc, v + 8 * half, 32);
      acc[0][v] *= scv; acc[1][v] *= scv; acc[2][v] *= scv; acc[3][v] *= scv;
    }

    // pack P tile (already in A-matrix order), read shared B tiles from LDS,
    // then run the 4 WMMAs
    v8u praw;
#pragma unroll
    for (int v = 0; v < 8; ++v) praw[v] = pk_bf16(p[2 * v], p[2 * v + 1]);
    v16bf a = __builtin_bit_cast(v16bf, praw);
    v8u braw[4];
#pragma unroll
    for (int c = 0; c < 4; ++c)
      braw[c] = *(const v8u*)&ldsB[cur][(c * 32 + lane) * 8];
#pragma unroll
    for (int c = 0; c < 4; ++c)
      acc[c] = __builtin_amdgcn_wmma_f32_16x16x32_bf16(
          false, a, false, __builtin_bit_cast(v16bf, braw[c]), (short)0, acc[c], false, false);

    // close the stage: my async portion landed, then block-wide barrier so
    // (a) next chunk is fully resident, (b) everyone is done reading `cur`
    // before the next iteration's async copy overwrites it
    wait_async0();
    __syncthreads();
  }

  // epilogue: divide by softmax sum, elu, store per-head
#pragma unroll
  for (int v = 0; v < 8; ++v) {
    float Sr = __shfl(S, v + 8 * half, 32);
    float inv = 1.f / Sr;
    int orow = ntile * 16 + v + 8 * half;
#pragma unroll
    for (int c = 0; c < 4; ++c) {
      float x = acc[c][v] * inv;
      x = (x > 0.f) ? x : expm1f(x);
      hprime[((size_t)head * NN + orow) * OO + lrow + 16 * c] = x;
    }
  }
}

// ---- 5. mean over heads ---------------------------------------------------
__global__ void __launch_bounds__(256)
gat_head_mean(const float* __restrict__ hprime, float* __restrict__ out) {
  const int t = blockIdx.x * blockDim.x + threadIdx.x;  // N*64
  float s = 0.f;
#pragma unroll
  for (int h = 0; h < HH; ++h) s += hprime[(size_t)h * NN * OO + t];
  out[t] = s * (1.0f / HH);
}

// ---------------------------------------------------------------------------
extern "C" void kernel_launch(void* const* d_in, const int* in_sizes, int n_in,
                              void* d_out, int out_size, void* d_ws, size_t ws_size,
                              hipStream_t stream) {
  const float* h   = (const float*)d_in[0];
  const int*   adj = (const int*)d_in[1];
  const float* W   = (const float*)d_in[2];
  const float* a1  = (const float*)d_in[3];
  const float* a2  = (const float*)d_in[4];
  float* out = (float*)d_out;

  char* ws = (char*)d_ws;
  float*        hw_f32 = (float*)(ws + OFF_HW);
  unsigned int* apack  = (unsigned int*)(ws + OFF_APACK);
  unsigned int* wpack  = (unsigned int*)(ws + OFF_WPACK);
  unsigned int* hwpack = (unsigned int*)(ws + OFF_HWPK);
  float*        f1     = (float*)(ws + OFF_F1);
  float*        f2     = (float*)(ws + OFF_F2);
  float*        hp     = (float*)(ws + OFF_HP);

  gat_pack_h   <<<dim3(FIN / 32, NN / 16), 32, 0, stream>>>(h, apack);
  gat_pack_w   <<<dim3((FIN / 32) * 4, HH), 32, 0, stream>>>(W, wpack);
  gat_hw_gemm  <<<dim3(NN / 16, HH), 32, 0, stream>>>(apack, wpack, hw_f32);
  gat_pack_hw  <<<dim3((NN / 32) * 4, HH), 32, 0, stream>>>(hw_f32, hwpack);
  gat_f1f2     <<<(HH * NN) / 256, 256, 0, stream>>>(hw_f32, a1, a2, f1, f2);
  gat_attn     <<<dim3(NN / 128, HH), 256, 0, stream>>>(f1, f2, adj, hwpack, hp);
  gat_head_mean<<<(NN * OO) / 256, 256, 0, stream>>>(hp, out);
}
